// Critic_72456098284041
// MI455X (gfx1250) — compile-verified, matrix operand order checked by reading, not run on hardware
//
#include <hip/hip_runtime.h>
#include <hip/hip_bf16.h>
#include <math.h>

// ---------------------------------------------------------------------------
// Critic forward for MI455X (gfx1250, wave32, WMMA), v2:
// all WMMA operands pre-packed into fragment-native layouts so every
// fragment load is b128-vectorized (global or LDS), no per-element gathers
// in the hot loops.
// ---------------------------------------------------------------------------

typedef _Float16 f16;
typedef __attribute__((ext_vector_type(16))) _Float16 v16h;
typedef __attribute__((ext_vector_type(8)))  _Float16 v8h;
typedef __attribute__((ext_vector_type(8)))  float    v8f;

#define B_   256
#define T_   512
#define A_   32
#define XW   160     // x width: 32 (act) + 64 (inp2) + 64 (inp3 const)
#define GW   1024    // 4 * U_LSTM
#define U_   256
#define NT_G 64      // GW/16 column tiles
#define KT_X 5       // XW/32 k tiles
#define KT_U 8       // U_/32 k tiles
#define ZPAD 1032    // padded z row stride (f32)
#define HP_SZ (KT_U * 512)                 // packed h: 8 tiles * 32 lanes * 16 halves
#define LSTM_LDS (HP_SZ * (int)sizeof(f16) + 16 * ZPAD * (int)sizeof(float))

__device__ __forceinline__ float elu1(float x) { return x > 0.f ? x : (__expf(x) - 1.f); }
__device__ __forceinline__ float sigm(float x) { return 1.f / (1.f + __expf(-x)); }

// ---- CDNA5 wave32 16x16x32 f16 fragment layouts (ISA 7.12.2) --------------
// A tile (16x32): lane L holds row M=L&15; lanes<16 hold K {0..7,16..23},
// lanes>=16 hold K {8..15,24..31}; element e -> k = kb + (e<8 ? e : e+8).
// B tile (32x16): same packing with N on lanes. C/D: lane L: N=L&15,
// M = v + (L>=16 ? 8 : 0) for VGPR v=0..7.

// generic (slow-path) gather loaders for the tiny phase-1 GEMMs
template <typename AT>
__device__ __forceinline__ v16h load_a_gather(const AT* A, int lda, int row0, int k0, int lane) {
    v16h a;
    int r  = lane & 15;
    int kb = (lane >> 4) ? 8 : 0;
    const AT* p = A + (size_t)(row0 + r) * lda + k0 + kb;
#pragma unroll
    for (int e = 0; e < 16; ++e) { int k = (e < 8) ? e : (e + 8); a[e] = (f16)p[k]; }
    return a;
}
template <typename WT>
__device__ __forceinline__ v16h load_b_gather(const WT* W, int ldw, int k0, int col0, int lane) {
    v16h b;
    int n  = lane & 15;
    int kb = (lane >> 4) ? 8 : 0;
    const WT* p = W + (size_t)(k0 + kb) * ldw + col0 + n;
#pragma unroll
    for (int e = 0; e < 16; ++e) { int k = (e < 8) ? e : (e + 8); b[e] = (f16)p[(size_t)k * ldw]; }
    return b;
}

// ---- generic small GEMM: out = act(A @ W + bias) --------------------------
template <typename AT, typename OT, bool DO_ELU>
__global__ void gemm_bias_act(const AT* __restrict__ A, int lda,
                              const float* __restrict__ W, int ldw,
                              const float* __restrict__ bias,
                              OT* __restrict__ out, int ldo,
                              int M, int N, int K) {
    int lane = threadIdx.x & 31;
    int wave = threadIdx.x >> 5;
    int col0 = (blockIdx.x * (blockDim.x >> 5) + wave) * 16;
    int row0 = blockIdx.y * 16;
    if (col0 >= N || row0 >= M) return;
    v8f acc = {};
    for (int k0 = 0; k0 < K; k0 += 32) {
        v16h a = load_a_gather(A, lda, row0, k0, lane);
        v16h b = load_b_gather(W, ldw, k0, col0, lane);
        acc = __builtin_amdgcn_wmma_f32_16x16x32_f16(false, a, false, b, (short)0, acc, false, false);
    }
    int n = lane & 15, mb = (lane >> 4) ? 8 : 0;
    float bv = bias ? bias[col0 + n] : 0.f;
#pragma unroll
    for (int v = 0; v < 8; ++v) {
        float x = acc[v] + bv;
        if (DO_ELU) x = elu1(x);
        out[(size_t)(row0 + v + mb) * ldo + col0 + n] = (OT)x;
    }
}

// ---- packers: row-major -> fragment-native layouts -------------------------
// B-pack: W f32 [K,N] -> P f16 [N/16][K/32][32 lanes][16], 32B/lane chunks
__global__ void pack_b(const float* __restrict__ W, f16* __restrict__ P, int K, int N) {
    int gid = blockIdx.x * blockDim.x + threadIdx.x;
    if (gid >= K * N) return;
    int e    = gid & 15;
    int lane = (gid >> 4) & 31;
    int nkt  = K >> 5;
    int kt   = (gid >> 9) % nkt;
    int nt   = gid / (512 * nkt);
    int n  = lane & 15;
    int kb = (lane >> 4) ? 8 : 0;
    int k  = kt * 32 + kb + ((e < 8) ? e : (e + 8));
    P[gid] = (f16)W[(size_t)k * N + nt * 16 + n];
}
// A-pack: x f16 [M,XW] -> P f16 [M/16][XW/32][32][16]
__global__ void pack_a(const f16* __restrict__ X, f16* __restrict__ P, int M) {
    int gid = blockIdx.x * blockDim.x + threadIdx.x;
    if (gid >= M * XW) return;
    int e    = gid & 15;
    int lane = (gid >> 4) & 31;
    int kt   = (gid >> 9) % KT_X;
    int mt   = gid / (512 * KT_X);
    int r  = lane & 15;
    int kb = (lane >> 4) ? 8 : 0;
    int k  = kt * 32 + kb + ((e < 8) ? e : (e + 8));
    P[gid] = X[(size_t)(mt * 16 + r) * XW + k];
}

// ---- x assembly: cols 0..31 = action*mu+mean ; cols 96..159 = elu(boi) ----
__global__ void prep_x(const float* __restrict__ action, const float* __restrict__ mu,
                       const float* __restrict__ mean, const float* __restrict__ boi,
                       f16* __restrict__ x) {
    int gid = blockIdx.x * blockDim.x + threadIdx.x;
    int m   = gid / 96;
    int idx = gid - m * 96;
    if (m >= B_ * T_) return;
    int b = m >> 9;
    float v; int col;
    if (idx < 32) {
        v = action[(size_t)m * A_ + idx] * mu[b * A_ + idx] + mean[b * A_ + idx];
        col = idx;
    } else {
        int j = idx - 32;
        v = elu1(boi[j]);                // faithful bug: inp3 = elu(bias) broadcast
        col = 96 + j;
    }
    x[(size_t)m * XW + col] = (f16)v;
}

// ---- x_proj: fully packed GEMM, stores straight into LSTM C-frag layout ---
// xpP index: ((rb*T + t)*64 + nt)*32*8 + lane_c*8 + v_c  (f16)
__global__ void gemm_xproj(const f16* __restrict__ xA, const f16* __restrict__ WlP,
                           const float* __restrict__ bl, f16* __restrict__ xpP) {
    int lane = threadIdx.x & 31, wave = threadIdx.x >> 5;
    int nt = blockIdx.x * 8 + wave;      // 0..63
    int mt = blockIdx.y;                 // 0..8191  (m = b*T + t rows)
    const f16* ap = xA  + ((size_t)mt * KT_X * 32 + lane) * 16;
    const f16* bp = WlP + ((size_t)nt * KT_X * 32 + lane) * 16;
    v8f acc = {};
#pragma unroll
    for (int kk = 0; kk < KT_X; ++kk) {
        v16h a = *(const v16h*)(ap + kk * 512);
        v16h b = *(const v16h*)(bp + kk * 512);
        acc = __builtin_amdgcn_wmma_f32_16x16x32_f16(false, a, false, b, (short)0, acc, false, false);
    }
    int n = lane & 15, mb = (lane >> 4) ? 8 : 0;
    int m0 = mt * 16;
    int b  = m0 >> 9;                    // whole tile shares one batch row
    int t0 = m0 & 511;
    int rb = b >> 4;
    int lane_c = n + ((b & 8) ? 16 : 0);
    int v_c    = b & 7;
    float bv = bl[nt * 16 + n];
#pragma unroll
    for (int v = 0; v < 8; ++v) {
        int t = t0 + v + mb;
        xpP[((((size_t)rb * T_ + t) * NT_G + nt) * 32 + lane_c) * 8 + v_c] = (f16)(acc[v] + bv);
    }
}

// ---- sequential LSTM + max-over-time + out projection ---------------------
// 16 blocks x 256 threads; block rb owns batch rows [16rb,16rb+16).
// h kept in LDS in packed A-fragment layout; all hot loads are b128.
__global__ void lstm_max_out(const f16* __restrict__ xpP,  // C-frag packed
                             const float* __restrict__ h0, // [256,256] (h0 == c0)
                             const f16* __restrict__ UlP,  // B-frag packed [64][8][32][16]
                             const float* __restrict__ Wo, // [256]
                             const float* __restrict__ bo, // [1]
                             float* __restrict__ out) {    // [256]
    extern __shared__ char smem[];
    f16*   hP  = (f16*)smem;                              // packed A frags, 8KB
    float* zS  = (float*)(smem + HP_SZ * sizeof(f16));    // [16][ZPAD]
    float* red = zS;

    int tid = threadIdx.x;
    int lane = tid & 31, wave = tid >> 5;
    int rb = blockIdx.x;

    // this thread owns hidden column j = tid; its packed-A slot:
    int jt = tid >> 5;                                    // K tile
    int lo = ((tid >> 3) & 1) ? 16 : 0;                   // lane-half offset
    int ee = (tid & 7) + (((tid >> 4) & 1) ? 8 : 0);      // element within v16h
    f16* hw = hP + jt * 512 + ee;

    float c[16], hmax[16];
#pragma unroll
    for (int r = 0; r < 16; ++r) {
        float v = h0[(size_t)(rb * 16 + r) * U_ + tid];
        c[r] = v;
        hmax[r] = -3.4e38f;                               // h0 not part of scan output
        hw[(r + lo) * 16] = (f16)v;
    }
    __syncthreads();

    for (int t = 0; t < T_; ++t) {
        // z = xp_t + h @ Ul : each wave does 8 column tiles of 16
        for (int tt = 0; tt < 8; ++tt) {
            int ct = wave * 8 + tt;
            // C init: one 16-byte vector load per lane (packed layout)
            v8h c8 = *(const v8h*)(xpP + ((((size_t)rb * T_ + t) * NT_G + ct) * 32 + lane) * 8);
            v8f acc;
#pragma unroll
            for (int v = 0; v < 8; ++v) acc[v] = (float)c8[v];
            const f16* bp = UlP + ((size_t)ct * KT_U * 32 + lane) * 16;
#pragma unroll
            for (int kk = 0; kk < KT_U; ++kk) {
                v16h a = *(const v16h*)(hP + (kk * 32 + lane) * 16);   // 2x ds_load_b128
                v16h b = *(const v16h*)(bp + kk * 512);                // 2x global b128
                acc = __builtin_amdgcn_wmma_f32_16x16x32_f16(false, a, false, b, (short)0, acc, false, false);
            }
            int n = lane & 15, mb = (lane >> 4) ? 8 : 0;
#pragma unroll
            for (int v = 0; v < 8; ++v)
                zS[(v + mb) * ZPAD + ct * 16 + n] = acc[v];
        }
        __syncthreads();

        if (t + 1 < T_) {  // next step's 32KB xp slice is contiguous: prefetch it
            const char* nxt = (const char*)(xpP + (((size_t)rb * T_ + t + 1) * NT_G) * 32 * 8);
            __builtin_prefetch(nxt + tid * 128, 0, 0);
        }

        // gates + state update (thread owns hidden col tid, rows 0..15)
#pragma unroll
        for (int r = 0; r < 16; ++r) {
            float zi = zS[r * ZPAD +        tid];
            float zf = zS[r * ZPAD +  256 + tid];
            float zg = zS[r * ZPAD +  512 + tid];
            float zo = zS[r * ZPAD +  768 + tid];
            float cc = sigm(zf) * c[r] + sigm(zi) * tanhf(zg);
            c[r] = cc;
            float h = sigm(zo) * tanhf(cc);
            hmax[r] = fmaxf(hmax[r], h);
            hw[(r + lo) * 16] = (f16)h;                   // packed-A store for next step
        }
        __syncthreads();
    }

    // out[row] = elu( sum_j hmax[row][j] * Wo[j] + bo )
    if (tid < 16) red[tid] = 0.f;
    __syncthreads();
    float w = Wo[tid];
#pragma unroll
    for (int r = 0; r < 16; ++r)
        atomicAdd(&red[r], hmax[r] * w);
    __syncthreads();
    if (tid < 16)
        out[rb * 16 + tid] = elu1(red[tid] + bo[0]);
}

extern "C" void kernel_launch(void* const* d_in, const int* in_sizes, int n_in,
                              void* d_out, int out_size, void* d_ws, size_t ws_size,
                              hipStream_t stream) {
    const float* motion = (const float*)d_in[0];
    const float* robot  = (const float*)d_in[1];
    /* d_in[2] osc_state: unused by forward */
    const float* action = (const float*)d_in[3];
    const float* osc    = (const float*)d_in[4];
    const float* mu     = (const float*)d_in[5];
    const float* mean   = (const float*)d_in[6];
    const float* Wm  = (const float*)d_in[7];  const float* bm  = (const float*)d_in[8];
    const float* Wr  = (const float*)d_in[9];  const float* br  = (const float*)d_in[10];
    const float* Wc  = (const float*)d_in[11]; const float* bc  = (const float*)d_in[12];
    const float* Wor = (const float*)d_in[13]; const float* bor = (const float*)d_in[14];
    /* d_in[15] Woi has 0 elements (faithful bug) */
    const float* boi = (const float*)d_in[16];
    const float* Wl  = (const float*)d_in[17]; const float* Ul  = (const float*)d_in[18];
    const float* bl  = (const float*)d_in[19];
    const float* Wo  = (const float*)d_in[20]; const float* bo  = (const float*)d_in[21];
    float* out = (float*)d_out;
    (void)in_sizes; (void)n_in; (void)out_size; (void)ws_size;

    char* ws = (char*)d_ws;
    size_t off = 0;
    auto alloc = [&](size_t bytes) -> void* {
        void* p = ws + off; off += (bytes + 255) & ~(size_t)255; return p;
    };
    f16*   x    = (f16*)  alloc((size_t)B_ * T_ * XW * sizeof(f16));   // row-major x
    f16*   xA   = (f16*)  alloc((size_t)B_ * T_ * XW * sizeof(f16));   // A-frag packed x
    f16*   xpP  = (f16*)  alloc((size_t)B_ * T_ * GW * sizeof(f16));   // C-frag packed x_proj
    f16*   WlP  = (f16*)  alloc((size_t)XW * GW * sizeof(f16));        // B-frag packed Wl
    f16*   UlP  = (f16*)  alloc((size_t)U_ * GW * sizeof(f16));        // B-frag packed Ul
    f16*   msrs = (f16*)  alloc((size_t)B_ * 512 * sizeof(f16));       // concat(ms,rs)
    float* h0   = (float*)alloc((size_t)B_ * U_ * sizeof(float));

    const int M = B_ * T_;

    // 1) assemble row-major x: act cols + constant elu(boi) cols
    prep_x<<<(M * 96 + 255) / 256, 256, 0, stream>>>(action, mu, mean, boi, x);
    // 2) inp2 = elu(osc[...,:64] @ Wor + bor) -> x[:,32:96]
    gemm_bias_act<float, f16, true><<<dim3(1, M / 16), 256, 0, stream>>>(
        osc, 128, Wor, 64, bor, x + 32, XW, M, 64, 64);
    // 3) pack operands into fragment-native layouts
    pack_a<<<(M * XW + 255) / 256, 256, 0, stream>>>(x, xA, M);
    pack_b<<<(XW * GW + 255) / 256, 256, 0, stream>>>(Wl, WlP, XW, GW);
    pack_b<<<(U_ * GW + 255) / 256, 256, 0, stream>>>(Ul, UlP, U_, GW);
    // 4) h0 = c0 = elu(concat(ms,rs) @ Wc + bc)
    gemm_bias_act<float, f16, true><<<dim3(2, 16), 256, 0, stream>>>(
        motion, 64, Wm, 256, bm, msrs, 512, B_, 256, 64);
    gemm_bias_act<float, f16, true><<<dim3(2, 16), 256, 0, stream>>>(
        robot, 128, Wr, 256, br, msrs + 256, 512, B_, 256, 128);
    gemm_bias_act<f16, float, true><<<dim3(2, 16), 256, 0, stream>>>(
        msrs, 512, Wc, 256, bc, h0, 256, B_, 256, 512);
    // 5) x_proj = x @ Wl + bl (packed in, packed out)
    gemm_xproj<<<dim3(8, M / 16), 256, 0, stream>>>(xA, WlP, bl, xpP);
    // 6) sequential LSTM + max + out projection
    hipFuncSetAttribute((const void*)lstm_max_out,
                        hipFuncAttributeMaxDynamicSharedMemorySize, LSTM_LDS);
    lstm_max_out<<<16, 256, LSTM_LDS, stream>>>(xpP, h0, UlP, Wo, bo, out);
}
